// VARNN_2345052143963
// MI455X (gfx1250) — matrix operand order repaired and checked by reading, not compiled, not run on hardware
//
#include <hip/hip_runtime.h>
#include <hip/hip_bf16.h>
#include <stdint.h>

// ---------------------------------------------------------------------------
// VARNN forward on MI455X (gfx1250): f16 WMMA (f32 accumulate) GEMM pipeline.
// Round 3: widen the betas-TVPL block N-footprint (12 coefficients) to cut
// L2 re-streaming of the 16MB activation matrix from 66x to 11x.
// ---------------------------------------------------------------------------

typedef __attribute__((ext_vector_type(16))) _Float16 v16h;
typedef __attribute__((ext_vector_type(8)))  _Float16 v8h;
typedef __attribute__((ext_vector_type(8)))  float    v8f;

#define B_ROWS 32768

// ---- WMMA fragment loaders -------------------------------------------------
// A matrix (16x32 f16, row-major [M,ld]):
//   lane L (g=L/16, m=L%16): elems e = A[m][k0+g*16+e], e = 0..15
__device__ __forceinline__ v16h load_a_frag(const _Float16* __restrict__ A,
                                            int row, int ld, int k0, int g) {
  return *(const v16h*)(A + (size_t)row * ld + k0 + g * 16);
}

// B matrix (32x16 f16), B[k][n] = W[n][k] with W row-major [N,ld]:
//   lane L (g=L/16, n=L%16): elems e = B[k0+g*16+e][n] = W[n][k0+g*16+e]
//   -> 16 contiguous halves, 32B aligned.
__device__ __forceinline__ v16h load_b_frag(const _Float16* __restrict__ W,
                                            int col, int ld, int k0, int g) {
  return *(const v16h*)(W + (size_t)col * ld + k0 + g * 16);
}

// ---- Trunk GEMM: Out[M,256] = relu(A[M,Kd] * W[256,Kd]^T + bias) -----------
// Block = 256 thr (8 waves): 2 waves in M x 4 waves in N.
// Wave tile 64x64: acc[4][4] v8f (128 VGPRs); 8 loads / 16 WMMAs per K-step.
__global__ __launch_bounds__(256) void gemm_relu_f16(
    const _Float16* __restrict__ A, const _Float16* __restrict__ W,
    const float* __restrict__ bias, _Float16* __restrict__ Out, int Kd) {
  const int lane = threadIdx.x & 31;
  const int w    = threadIdx.x >> 5;
  const int g    = lane >> 4;
  const int nl   = lane & 15;
  const int m0   = blockIdx.x * 128 + (w & 1) * 64;  // 64 rows per wave
  const int n0   = (w >> 1) * 64;                    // 64 cols per wave

  const v8f vz = {0.f, 0.f, 0.f, 0.f, 0.f, 0.f, 0.f, 0.f};
  v8f acc[4][4];
#pragma unroll
  for (int t = 0; t < 4; ++t)
#pragma unroll
    for (int s = 0; s < 4; ++s) acc[t][s] = vz;

  for (int k0 = 0; k0 < Kd; k0 += 32) {
    v16h bf[4];
#pragma unroll
    for (int s = 0; s < 4; ++s)
      bf[s] = load_b_frag(W, n0 + s * 16 + nl, Kd, k0, g);
#pragma unroll
    for (int t = 0; t < 4; ++t) {
      v16h a = load_a_frag(A, m0 + t * 16 + nl, Kd, k0, g);
#pragma unroll
      for (int s = 0; s < 4; ++s)
        acc[t][s] = __builtin_amdgcn_wmma_f32_16x16x32_f16(
            false, a, false, bf[s], (short)0, acc[t][s], false, false);
    }
  }
  // C layout: vgpr r -> row = g*8 + r, col = nl (per 16-col subtile)
#pragma unroll
  for (int t = 0; t < 4; ++t) {
#pragma unroll
    for (int s = 0; s < 4; ++s) {
      const int n = n0 + s * 16 + nl;
      const float bz = bias[n];
#pragma unroll
      for (int r = 0; r < 8; ++r) {
        float v = acc[t][s][r] + bz;
        v = v > 0.f ? v : 0.f;
        Out[(size_t)(m0 + t * 16 + g * 8 + r) * 256 + n] = (_Float16)v;
      }
    }
  }
}

// ---- TVPL head (wide-N): Out[b,k] = W_out[k].relu(H W_tvpl[k]^T + b_tvpl[k]) + b_out[k]
// Block = 8 waves: 2 M-waves (64 rows each) x 4 N-waves (NC coefficients each).
// Block tile: 128 rows x 4*NC coefficients. Grid = (B/128, Kc/(4*NC)).
template <int NC>
__global__ __launch_bounds__(256) void tvpl_head_wide(
    const _Float16* __restrict__ H,   // [M,256] f16
    const _Float16* __restrict__ Wt,  // [Kc*16,256] f16
    const float* __restrict__ bt,     // [Kc*16]
    const float* __restrict__ wo,     // [Kc*16]  (W_out flattened)
    const float* __restrict__ bo,     // [Kc]
    float* __restrict__ Out,          // [M,Kc]
    int Kc) {
  const int lane  = threadIdx.x & 31;
  const int w     = threadIdx.x >> 5;
  const int g     = lane >> 4;
  const int nl    = lane & 15;
  const int m0    = blockIdx.x * 128 + (w & 1) * 64;
  const int kbase = (blockIdx.y * 4 + (w >> 1)) * NC;  // NC coefs per N-wave

  const v8f vz = {0.f, 0.f, 0.f, 0.f, 0.f, 0.f, 0.f, 0.f};
  v8f acc[4][NC];
#pragma unroll
  for (int t = 0; t < 4; ++t)
#pragma unroll
    for (int j = 0; j < NC; ++j) acc[t][j] = vz;

#pragma unroll
  for (int k0 = 0; k0 < 256; k0 += 32) {
    v16h bf[NC];
#pragma unroll
    for (int j = 0; j < NC; ++j)
      bf[j] = load_b_frag(Wt, (kbase + j) * 16 + nl, 256, k0, g);
#pragma unroll
    for (int t = 0; t < 4; ++t) {
      v16h a = load_a_frag(H, m0 + t * 16 + nl, 256, k0, g);
#pragma unroll
      for (int j = 0; j < NC; ++j)
        acc[t][j] = __builtin_amdgcn_wmma_f32_16x16x32_f16(
            false, a, false, bf[j], (short)0, acc[t][j], false, false);
    }
  }
#pragma unroll
  for (int j = 0; j < NC; ++j) {
    const int k = kbase + j;
    const float bz  = bt[k * 16 + nl];
    const float wz  = wo[k * 16 + nl];
    const float bok = bo[k];
#pragma unroll
    for (int t = 0; t < 4; ++t) {
#pragma unroll
      for (int r = 0; r < 8; ++r) {
        float v = acc[t][j][r] + bz;
        v = v > 0.f ? v : 0.f;
        v *= wz;
#pragma unroll
        for (int off = 8; off >= 1; off >>= 1) v += __shfl_xor(v, off, 16);
        if (nl == 0)
          Out[(size_t)(m0 + t * 16 + g * 8 + r) * Kc + k] = v + bok;
      }
    }
  }
}

// ---- TVPL head (narrow, for Kc=10): 8 waves stacked in M, 2 coefs/block ----
// Grid = (B/512, Kc/2).
__global__ __launch_bounds__(256) void tvpl_head_f16(
    const _Float16* __restrict__ H, const _Float16* __restrict__ Wt,
    const float* __restrict__ bt, const float* __restrict__ wo,
    const float* __restrict__ bo, float* __restrict__ Out, int Kc) {
  const int lane = threadIdx.x & 31;
  const int w    = threadIdx.x >> 5;
  const int g    = lane >> 4;
  const int nl   = lane & 15;
  const int k2   = blockIdx.y * 2;
  const int m0   = blockIdx.x * 512 + w * 64;

  const v8f vz = {0.f, 0.f, 0.f, 0.f, 0.f, 0.f, 0.f, 0.f};
  v8f acc[4][2];
#pragma unroll
  for (int t = 0; t < 4; ++t) { acc[t][0] = vz; acc[t][1] = vz; }

#pragma unroll
  for (int k0 = 0; k0 < 256; k0 += 32) {
    v16h bf[2];
#pragma unroll
    for (int j = 0; j < 2; ++j)
      bf[j] = load_b_frag(Wt, (k2 + j) * 16 + nl, 256, k0, g);
#pragma unroll
    for (int t = 0; t < 4; ++t) {
      v16h a = load_a_frag(H, m0 + t * 16 + nl, 256, k0, g);
#pragma unroll
      for (int j = 0; j < 2; ++j)
        acc[t][j] = __builtin_amdgcn_wmma_f32_16x16x32_f16(
            false, a, false, bf[j], (short)0, acc[t][j], false, false);
    }
  }
#pragma unroll
  for (int j = 0; j < 2; ++j) {
    const int k = k2 + j;
    const float bz  = bt[k * 16 + nl];
    const float wz  = wo[k * 16 + nl];
    const float bok = bo[k];
#pragma unroll
    for (int t = 0; t < 4; ++t) {
#pragma unroll
      for (int r = 0; r < 8; ++r) {
        float v = acc[t][j][r] + bz;
        v = v > 0.f ? v : 0.f;
        v *= wz;
#pragma unroll
        for (int off = 8; off >= 1; off >>= 1) v += __shfl_xor(v, off, 16);
        if (nl == 0)
          Out[(size_t)(m0 + t * 16 + g * 8 + r) * Kc + k] = v + bok;
      }
    }
  }
}

// ---- fp32 -> f16 convert with optional K-padding ---------------------------
__global__ void k_cvt_pad(const float* __restrict__ src, _Float16* __restrict__ dst,
                          int rows, int src_cols, int dst_cols) {
  int i = blockIdx.x * blockDim.x + threadIdx.x;
  if (i >= rows * dst_cols) return;
  int r = i / dst_cols, c = i - r * dst_cols;
  float v = (c < src_cols) ? src[(size_t)r * src_cols + c] : 0.f;
  dst[i] = (_Float16)v;
}

// ---- Build branch inputs: xh0 = S[:,0:32]/sqrt(32), xh1 = pad32(S[:,32]) ---
__global__ void k_build_x(const float* __restrict__ S,
                          _Float16* __restrict__ xh0, _Float16* __restrict__ xh1) {
  int i = blockIdx.x * blockDim.x + threadIdx.x;  // over B*32
  if (i >= B_ROWS * 32) return;
  int b = i >> 5, c = i & 31;
  xh0[i] = (_Float16)(S[(size_t)b * 33 + c] * 0.17677669529663687f);
  xh1[i] = (c == 0) ? (_Float16)S[(size_t)b * 33 + 32] : (_Float16)0.f;
}

// ---- Finalize: y_hat, precision, stacked betas/chol ------------------------
__global__ void k_finalize(const float* __restrict__ S,
                           const float* __restrict__ oB0, const float* __restrict__ oB1,
                           const float* __restrict__ oA0, const float* __restrict__ oA1,
                           float* __restrict__ out, int B) {
  int b = blockIdx.x * blockDim.x + threadIdx.x;
  if (b >= B) return;
  const float* p0 = oB0 + (size_t)b * 132;  // [4,33] row-major
  const float* p1 = oB1 + (size_t)b * 132;
  const float* srow = S + (size_t)b * 33;

  // y_hat[o] = betas_red[o][0] + sum_d betas_red[o][1+d] * S[b,d]
  for (int o = 0; o < 4; ++o) {
    float acc = p0[o * 33] + p1[o * 33];
    for (int d = 0; d < 32; ++d)
      acc += (p0[o * 33 + 1 + d] + p1[o * 33 + 1 + d]) * srow[d];
    out[(size_t)b * 4 + o] = acc;
  }

  // Cholesky factors (lower-tri, abs on diagonal)
  const int rr[10] = {0, 1, 1, 2, 2, 2, 3, 3, 3, 3};
  const int cc[10] = {0, 0, 1, 0, 1, 2, 0, 1, 2, 3};
  float c0[16], c1[16];
  for (int i = 0; i < 16; ++i) { c0[i] = 0.f; c1[i] = 0.f; }
  for (int t = 0; t < 10; ++t) {
    float v0 = oA0[(size_t)b * 10 + t];
    float v1 = oA1[(size_t)b * 10 + t];
    if (rr[t] == cc[t]) { v0 = fabsf(v0); v1 = fabsf(v1); }
    c0[rr[t] * 4 + cc[t]] = v0;
    c1[rr[t] * 4 + cc[t]] = v1;
  }
  float cr[16];
  for (int i = 0; i < 16; ++i) cr[i] = c0[i] + c1[i];

  // precision = chol_red * chol_red^T
  float* pout = out + (size_t)B * 4;
  for (int i = 0; i < 4; ++i)
    for (int j = 0; j < 4; ++j) {
      float a = 0.f;
      for (int t = 0; t < 4; ++t) a += cr[i * 4 + t] * cr[j * 4 + t];
      pout[(size_t)b * 16 + i * 4 + j] = a;
    }

  // betas_stacked [B,33,4,2]
  float* bst = out + (size_t)B * 20;
  for (int c = 0; c < 33; ++c)
    for (int o = 0; o < 4; ++o) {
      size_t idx = (((size_t)b * 33 + c) * 4 + o) * 2;
      bst[idx]     = p0[o * 33 + c];
      bst[idx + 1] = p1[o * 33 + c];
    }
  // chol_stacked [B,4,4,2]
  float* cst = out + (size_t)B * 284;
  for (int i = 0; i < 16; ++i) {
    cst[((size_t)b * 16 + i) * 2]     = c0[i];
    cst[((size_t)b * 16 + i) * 2 + 1] = c1[i];
  }
}

// ---------------------------------------------------------------------------
extern "C" void kernel_launch(void* const* d_in, const int* in_sizes, int n_in,
                              void* d_out, int out_size, void* d_ws, size_t ws_size,
                              hipStream_t stream) {
  (void)in_sizes; (void)n_in; (void)out_size; (void)ws_size;
  const int B = B_ROWS;
  const float* S = (const float*)d_in[0];

  // workspace carving (256B aligned chunks)
  uintptr_t p = (uintptr_t)d_ws;
  auto carve = [&](size_t bytes) -> void* {
    p = (p + 255) & ~(uintptr_t)255;
    void* r = (void*)p;
    p += bytes;
    return r;
  };
  _Float16* xh0  = (_Float16*)carve((size_t)B * 32 * 2);
  _Float16* xh1  = (_Float16*)carve((size_t)B * 32 * 2);
  _Float16* actA = (_Float16*)carve((size_t)B * 256 * 2);
  _Float16* actB = (_Float16*)carve((size_t)B * 256 * 2);

  // branch order in params dict: h0_betas, h0_alphas, h1_betas, h1_alphas
  const int Ks[4] = {132, 10, 132, 10};
  _Float16 *w_in16[4], *w_first16[4], *w_hid16[4], *w_tvpl16[4];
  float* outbuf[4];
  for (int br = 0; br < 4; ++br) {
    w_in16[br]    = (_Float16*)carve((size_t)256 * 32 * 2);
    w_first16[br] = (_Float16*)carve((size_t)256 * 256 * 2);
    w_hid16[br]   = (_Float16*)carve((size_t)256 * 256 * 2);
    w_tvpl16[br]  = (_Float16*)carve((size_t)Ks[br] * 16 * 256 * 2);
    outbuf[br]    = (float*)carve((size_t)B * Ks[br] * 4);
  }

  k_build_x<<<(B * 32 + 255) / 256, 256, 0, stream>>>(S, xh0, xh1);

  for (int br = 0; br < 4; ++br) {
    const float* const* P = (const float* const*)(d_in + 1 + br * 10);
    const float* W_in    = P[0];
    const float* b_in    = P[1];
    const float* W_first = P[2];
    const float* b_first = P[3];
    const float* W_hid   = P[4];
    const float* b_hid   = P[5];
    const float* W_tvpl  = P[6];
    const float* b_tvpl  = P[7];
    const float* W_out   = P[8];
    const float* b_out   = P[9];
    const int K = Ks[br];
    const int in_dim = (br < 2) ? 32 : 1;  // h1 pads [256,1] -> [256,32]

    int n = 256 * 32;
    k_cvt_pad<<<(n + 255) / 256, 256, 0, stream>>>(W_in, w_in16[br], 256, in_dim, 32);
    n = 256 * 256;
    k_cvt_pad<<<(n + 255) / 256, 256, 0, stream>>>(W_first, w_first16[br], 256, 256, 256);
    k_cvt_pad<<<(n + 255) / 256, 256, 0, stream>>>(W_hid, w_hid16[br], 256, 256, 256);
    n = K * 16 * 256;
    k_cvt_pad<<<(n + 255) / 256, 256, 0, stream>>>(W_tvpl, w_tvpl16[br], K * 16, 256, 256);

    const _Float16* X = (br < 2) ? xh0 : xh1;
    gemm_relu_f16<<<B / 128, 256, 0, stream>>>(X,    w_in16[br],    b_in,    actA, 32);
    gemm_relu_f16<<<B / 128, 256, 0, stream>>>(actA, w_first16[br], b_first, actB, 256);
    gemm_relu_f16<<<B / 128, 256, 0, stream>>>(actB, w_hid16[br],   b_hid,   actA, 256);

    if (K == 132) {
      // block = 128 rows x 12 coefs; grid (256, 11): A re-streamed 11x not 66x
      tvpl_head_wide<3><<<dim3(B / 128, K / 12), 256, 0, stream>>>(
          actA, w_tvpl16[br], b_tvpl, W_out, b_out, outbuf[br], K);
    } else {
      tvpl_head_f16<<<dim3(B / 512, K / 2), 256, 0, stream>>>(
          actA, w_tvpl16[br], b_tvpl, W_out, b_out, outbuf[br], K);
    }
  }

  k_finalize<<<(B + 255) / 256, 256, 0, stream>>>(
      S, outbuf[0], outbuf[2], outbuf[1], outbuf[3], (float*)d_out, B);
}
